// NodeEncoder_68693706932376
// MI455X (gfx1250) — matrix operand order repaired and checked by reading, not compiled, not run on hardware
//
#include <hip/hip_runtime.h>
#include <hip/hip_bf16.h>
#include <math.h>

typedef __bf16 bf16_t;
typedef __attribute__((ext_vector_type(16))) __bf16 v16bf;
typedef __attribute__((ext_vector_type(8)))  __bf16 v8bf;
typedef __attribute__((ext_vector_type(4)))  __bf16 v4bf;
typedef __attribute__((ext_vector_type(8)))  float  v8f;

#define H1F   256
#define H2F   128
#define OUTF  128
#define NEG_SLOPE 0.2f

// ---------------------------------------------------------------- utilities
__global__ void k_fill(float* __restrict__ p, long n, float v) {
  long i = (long)blockIdx.x * blockDim.x + threadIdx.x;
  long stride = (long)gridDim.x * blockDim.x;
  for (; i < n; i += stride) p[i] = v;
}

__device__ inline float atomic_max_f32(float* addr, float v) {
  if (v >= 0.0f)
    return __int_as_float(atomicMax((int*)addr, __float_as_int(v)));
  return __uint_as_float(atomicMin((unsigned int*)addr, __float_as_uint(v)));
}

// cs = W1 . att_src ; cd = W1 . att_dst   (rank-1 GAT scalars)
__global__ void k_scalars(const float* __restrict__ W1,
                          const float* __restrict__ as_,
                          const float* __restrict__ ad_,
                          float* __restrict__ scal) {
  __shared__ float ss[256];
  __shared__ float sd[256];
  int j = threadIdx.x;
  ss[j] = W1[j] * as_[j];
  sd[j] = W1[j] * ad_[j];
  __syncthreads();
  for (int o = 128; o > 0; o >>= 1) {
    if (j < o) { ss[j] += ss[j + o]; sd[j] += sd[j + o]; }
    __syncthreads();
  }
  if (j == 0) { scal[0] = ss[0]; scal[1] = sd[0]; }
}

// Pack concatenated bf16 weights: B1 = [Wl1;Wr1] (512x128), B2 = [Wl2;Wr2] (256x128)
__global__ void k_weights(const float* __restrict__ Wl1, const float* __restrict__ Wr1,
                          const float* __restrict__ Wl2, const float* __restrict__ Wr2,
                          bf16_t* __restrict__ B1, bf16_t* __restrict__ B2) {
  const int total1 = 2 * H1F * OUTF;     // 65536
  const int total2 = 2 * H2F * OUTF;     // 32768
  int i = blockIdx.x * blockDim.x + threadIdx.x;
  int stride = gridDim.x * blockDim.x;
  for (int t = i; t < total1 + total2; t += stride) {
    if (t < total1) {
      int k = t >> 7, n = t & 127;
      float v = (k < H1F) ? Wl1[k * OUTF + n] : Wr1[(k - H1F) * OUTF + n];
      B1[t] = (bf16_t)v;
    } else {
      int t2 = t - total1;
      int k = t2 >> 7, n = t2 & 127;
      float v = (k < H2F) ? Wl2[k * OUTF + n] : Wr2[(k - H2F) * OUTF + n];
      B2[t2] = (bf16_t)v;
    }
  }
}

// ---------------------------------------------------------------- GAT (scalar) passes
__global__ void k_deg(const int* __restrict__ dst, int E, float* __restrict__ deg) {
  int i = blockIdx.x * blockDim.x + threadIdx.x;
  if (i < E) atomicAdd(&deg[dst[i]], 1.0f);
}

__global__ void k_edge_max(const int* __restrict__ src, const int* __restrict__ dst,
                           const float* __restrict__ x, const float* __restrict__ scal,
                           int E, int N, float* __restrict__ m) {
  int i = blockIdx.x * blockDim.x + threadIdx.x;
  if (i >= E + N) return;
  int s, d;
  if (i < E) { s = src[i]; d = dst[i]; } else { s = d = i - E; }   // self loops
  float v = x[s] * scal[0] + x[d] * scal[1];
  v = (v > 0.0f) ? v : NEG_SLOPE * v;
  atomic_max_f32(&m[d], v);
}

__global__ void k_edge_sum(const int* __restrict__ src, const int* __restrict__ dst,
                           const float* __restrict__ x, const float* __restrict__ scal,
                           const float* __restrict__ m, int E, int N,
                           float* __restrict__ sb, float* __restrict__ nb) {
  int i = blockIdx.x * blockDim.x + threadIdx.x;
  if (i >= E + N) return;
  int s, d;
  if (i < E) { s = src[i]; d = dst[i]; } else { s = d = i - E; }
  float v = x[s] * scal[0] + x[d] * scal[1];
  v = (v > 0.0f) ? v : NEG_SLOPE * v;
  float e = __expf(v - m[d]);
  atomicAdd(&sb[d], e);
  atomicAdd(&nb[d], e * x[s]);
}

// h1[n,j] = relu(t[n]*W1[j] + b1[j]),  t[n] = num[n]/s[n]   (stored bf16)
__global__ void k_h1(const float* __restrict__ num, const float* __restrict__ s,
                     const float* __restrict__ W1, const float* __restrict__ b1,
                     bf16_t* __restrict__ h1, int N) {
  long i = (long)blockIdx.x * blockDim.x + threadIdx.x;
  if (i >= (long)N * H1F) return;
  int n = (int)(i >> 8);
  int j = (int)(i & 255);
  float t = num[n] / s[n];
  float v = fmaf(t, W1[j], b1[j]);
  h1[i] = (bf16_t)fmaxf(v, 0.0f);
}

__global__ void k_dinv(const float* __restrict__ deg, float* __restrict__ dinv, int N) {
  int i = blockIdx.x * blockDim.x + threadIdx.x;
  if (i < N) dinv[i] = 1.0f / fmaxf(deg[i], 1.0f);
}

// ---------------------------------------------------------------- edge aggregation
// warp-per-edge: lane handles 8 contiguous bf16 features (16B vector load)
__global__ void k_agg256(const int* __restrict__ src, const int* __restrict__ dst,
                         const bf16_t* __restrict__ h, float* __restrict__ agg, int E) {
  int gid = blockIdx.x * blockDim.x + threadIdx.x;
  int e = gid >> 5;
  int lane = gid & 31;
  if (e >= E) return;
  int s = src[e], d = dst[e];
  v8bf hv = *(const v8bf*)(h + (size_t)s * H1F + lane * 8);
  float* ap = agg + (size_t)d * H1F + lane * 8;
#pragma unroll
  for (int q = 0; q < 8; ++q) atomicAdd(ap + q, (float)hv[q]);
}

__global__ void k_agg128(const int* __restrict__ src, const int* __restrict__ dst,
                         const bf16_t* __restrict__ h, float* __restrict__ agg, int E) {
  int gid = blockIdx.x * blockDim.x + threadIdx.x;
  int e = gid >> 5;
  int lane = gid & 31;
  if (e >= E) return;
  int s = src[e], d = dst[e];
  v4bf hv = *(const v4bf*)(h + (size_t)s * H2F + lane * 4);
  float* ap = agg + (size_t)d * H2F + lane * 4;
#pragma unroll
  for (int q = 0; q < 4; ++q) atomicAdd(ap + q, (float)hv[q]);
}

// ---------------------------------------------------------------- fused SAGE GEMM
// out[N,128] = epilogue( [agg/deg | h] @ Bw  + bias )
// Block: 256 threads (8 wave32), tile 128(M) x 128(N), K chunked by 32 through LDS.
template <int KTOT, bool RELU, typename OutT>
__global__ void __launch_bounds__(256)
k_gemm(const float* __restrict__ agg, const bf16_t* __restrict__ h,
       const float* __restrict__ dinv, const bf16_t* __restrict__ Bw,
       const float* __restrict__ bias, OutT* __restrict__ out, int nrows) {
  constexpr int KA = KTOT / 2;      // width of agg (== width of h)
  constexpr int LD = 34;            // padded LDS row stride (17 dwords, conflict-free)
  __shared__ bf16_t la[128 * LD];   // A tile: [row][k]
  __shared__ bf16_t lb[128 * LD];   // B tile: [n][k]  (transposed on fill)

  const int tid  = threadIdx.x;
  const int wave = tid >> 5;
  const int lane = tid & 31;
  const int r0   = blockIdx.x * 128;

  // ISA 7.12.2 fragment geometry (wave32, 16-bit A 16x32 / B 32x16)
  const int mrow = (wave << 4) + (lane & 15);      // A row within block tile
  const int koffA = (lane < 16) ? 0 : 8;           // A: lanes 0-15 K{0..7,16..23}, 16-31 K{8..15,24..31}
  const int koffB = (lane < 16) ? 0 : 8;           // B: lanes 0-15 K{0..7,16..23}, 16-31 K{8..15,24..31}

  v8f acc[8] = {};

  for (int kc = 0; kc < KTOT; kc += 32) {
    // ---- stage A: 128 rows x 32 k; fuse mean-normalization + fp32->bf16
#pragma unroll
    for (int it = 0; it < 16; ++it) {
      int i  = tid + it * 256;
      int r  = i >> 5;
      int kk = i & 31;
      int row = r0 + r;
      int gk  = kc + kk;
      float v = 0.0f;
      if (row < nrows) {
        if (gk < KA) v = agg[(size_t)row * KA + gk] * dinv[row];
        else         v = (float)h[(size_t)row * KA + (gk - KA)];
      }
      la[r * LD + kk] = (bf16_t)v;
    }
    // ---- stage B: 32 k x 128 n, transposed into [n][k]
#pragma unroll
    for (int it = 0; it < 16; ++it) {
      int i  = tid + it * 256;
      int kk = i >> 7;
      int n  = i & 127;
      lb[n * LD + kk] = Bw[(size_t)(kc + kk) * OUTF + n];
    }
    __syncthreads();

    // ---- A fragment (reused across all 8 N-tiles)
    v16bf af;
#pragma unroll
    for (int p = 0; p < 8; ++p) {
      int k = (p < 4) ? (koffA + 2 * p) : (koffA + 16 + 2 * (p - 4));
      af[2 * p]     = la[mrow * LD + k];
      af[2 * p + 1] = la[mrow * LD + k + 1];
    }
#pragma unroll
    for (int nt = 0; nt < 8; ++nt) {
      v16bf bfrag;
      int n = (nt << 4) + (lane & 15);
#pragma unroll
      for (int p = 0; p < 8; ++p) {
        int k = (p < 4) ? (koffB + 2 * p) : (koffB + 16 + 2 * (p - 4));
        bfrag[2 * p]     = lb[n * LD + k];
        bfrag[2 * p + 1] = lb[n * LD + k + 1];
      }
      acc[nt] = __builtin_amdgcn_wmma_f32_16x16x32_bf16(
          /*neg_a=*/false, af, /*neg_b=*/false, bfrag,
          /*c_mod=*/(short)0, acc[nt], /*reuse_a=*/false, /*reuse_b=*/false);
    }
    __syncthreads();
  }

  // ---- epilogue: C/D layout (VGPR i -> M=i for lanes<16, M=8+i for lanes>=16)
  const int rhalf = (lane >> 4) * 8;
#pragma unroll
  for (int nt = 0; nt < 8; ++nt) {
    int col = (nt << 4) + (lane & 15);
    float b = bias[col];
#pragma unroll
    for (int i2 = 0; i2 < 8; ++i2) {
      int row = r0 + (wave << 4) + rhalf + i2;
      if (row < nrows) {
        float v = acc[nt][i2] + b;
        if (RELU) v = fmaxf(v, 0.0f);
        out[(size_t)row * OUTF + col] = (OutT)v;
      }
    }
  }
}

// ---------------------------------------------------------------- launch
extern "C" void kernel_launch(void* const* d_in, const int* in_sizes, int n_in,
                              void* d_out, int out_size, void* d_ws, size_t ws_size,
                              hipStream_t stream) {
  const float* x       = (const float*)d_in[0];
  const int*   ei      = (const int*)d_in[1];
  const float* W1      = (const float*)d_in[2];
  const float* att_src = (const float*)d_in[3];
  const float* att_dst = (const float*)d_in[4];
  const float* b1      = (const float*)d_in[5];
  const float* Wl1     = (const float*)d_in[6];
  const float* bl1     = (const float*)d_in[7];
  const float* Wr1     = (const float*)d_in[8];
  const float* Wl2     = (const float*)d_in[9];
  const float* bl2     = (const float*)d_in[10];
  const float* Wr2     = (const float*)d_in[11];

  const int N = in_sizes[0];
  const int E = in_sizes[1] / 2;
  const int* src = ei;
  const int* dst = ei + E;

  // workspace carve (256B aligned)
  char* w = (char*)d_ws;
  auto carve = [&](size_t bytes) {
    char* p = w;
    w += (bytes + 255) & ~(size_t)255;
    return p;
  };
  float*  deg  = (float*)carve((size_t)N * 4);
  float*  mbuf = (float*)carve((size_t)N * 4);
  float*  sbuf = (float*)carve((size_t)N * 4);
  float*  nbuf = (float*)carve((size_t)N * 4);
  float*  dinv = (float*)carve((size_t)N * 4);
  float*  scal = (float*)carve(256);
  bf16_t* h1   = (bf16_t*)carve((size_t)N * H1F * 2);
  float*  agg1 = (float*)carve((size_t)N * H1F * 4);
  bf16_t* h2   = (bf16_t*)carve((size_t)N * H2F * 2);
  float*  agg2 = (float*)carve((size_t)N * H2F * 4);
  bf16_t* B1   = (bf16_t*)carve((size_t)2 * H1F * OUTF * 2);
  bf16_t* B2   = (bf16_t*)carve((size_t)2 * H2F * OUTF * 2);

  const int TB = 256;
  // zero / init accumulators (must run every call: graph replay, poisoned ws)
  k_fill<<<1024, TB, 0, stream>>>(deg,  N, 0.0f);
  k_fill<<<1024, TB, 0, stream>>>(mbuf, N, -INFINITY);
  k_fill<<<1024, TB, 0, stream>>>(sbuf, N, 0.0f);
  k_fill<<<1024, TB, 0, stream>>>(nbuf, N, 0.0f);
  k_fill<<<4096, TB, 0, stream>>>(agg1, (long)N * H1F, 0.0f);
  k_fill<<<4096, TB, 0, stream>>>(agg2, (long)N * H2F, 0.0f);

  k_scalars<<<1, 256, 0, stream>>>(W1, att_src, att_dst, scal);
  k_weights<<<192, TB, 0, stream>>>(Wl1, Wr1, Wl2, Wr2, B1, B2);

  // GAT layer: scalar edge passes (rank-1 collapse)
  k_deg<<<(E + TB - 1) / TB, TB, 0, stream>>>(dst, E, deg);
  int en_blocks = (E + N + TB - 1) / TB;
  k_edge_max<<<en_blocks, TB, 0, stream>>>(src, dst, x, scal, E, N, mbuf);
  k_edge_sum<<<en_blocks, TB, 0, stream>>>(src, dst, x, scal, mbuf, E, N, sbuf, nbuf);
  k_h1<<<(int)(((long)N * H1F + TB - 1) / TB), TB, 0, stream>>>(nbuf, sbuf, W1, b1, h1, N);
  k_dinv<<<(N + TB - 1) / TB, TB, 0, stream>>>(deg, dinv, N);

  // SAGE layer 1: edge aggregation (warp/edge) + fused WMMA GEMM [N,512]x[512,128]
  long warps1 = (long)E * 32;
  k_agg256<<<(int)((warps1 + TB - 1) / TB), TB, 0, stream>>>(src, dst, h1, agg1, E);
  int gblocks = (N + 127) / 128;
  k_gemm<2 * H1F, true, bf16_t><<<gblocks, TB, 0, stream>>>(agg1, h1, dinv, B1, bl1, h2, N);

  // SAGE layer 2: aggregation + fused WMMA GEMM [N,256]x[256,128] -> fp32 out
  k_agg128<<<(int)((warps1 + TB - 1) / TB), TB, 0, stream>>>(src, dst, h2, agg2, E);
  k_gemm<2 * H2F, false, float><<<gblocks, TB, 0, stream>>>(agg2, h2, dinv, B2, bl2,
                                                            (float*)d_out, N);
}
